// DySample_1288490189337
// MI455X (gfx1250) — matrix operand order repaired and checked by reading, not compiled, not run on hardware
//
#include <hip/hip_runtime.h>

typedef __attribute__((ext_vector_type(16))) _Float16 v16h;
typedef __attribute__((ext_vector_type(8)))  float    v8f;

#define B_  16
#define C_  64
#define H_  128
#define W_  128
#define HW  (H_ * W_)          // 16384
#define OC  32
#define OH_ (2 * H_)           // 256
#define OW_ (2 * W_)           // 256
#define OHW ((size_t)OH_ * OW_)

union AFrag { _Float16 h[16]; v16h v; };
union CFrag { float f[8]; v8f v; };

// ---------------------------------------------------------------------------
// Kernel 1: off[b,o,h,w] = sum_c x[b,c,h,w] * w_off[o,c] + b_off[o]
// GEMM: M = B*HW pixels, N = 32 out-channels, K = 64 in-channels.
// One wave handles a 16-pixel M-tile x N=32 (two 16x16 accumulators),
// K=64 as two k-steps of v_wmma_f32_16x16x32_f16.
// ---------------------------------------------------------------------------
__global__ __launch_bounds__(256)
void dysample_offset_wmma(const float* __restrict__ x,
                          const float* __restrict__ w_off,
                          const float* __restrict__ b_off,
                          float* __restrict__ off)
{
    const int lane  = threadIdx.x & 31;
    const int wave  = threadIdx.x >> 5;
    const int tile  = blockIdx.x * 8 + wave;       // 16384 tiles total
    const int Mbase = tile * 16;                   // first pixel of tile
    const int b     = Mbase / HW;                  // whole tile in one batch
    const int pbase = Mbase % HW;

    const int hiHalf = (lane >= 16) ? 1 : 0;

    // ---- B fragments (loop-invariant): B[k][n] = w_off[n][k] ----
    // Layout (16-bit B, 32x16): N = lane&15, K = 16*(e>=8) + 8*(lane>=16) + (e&7)
    v16h bf[2][2];   // [kstep][ntile]
    {
        const int n  = lane & 15;
        const int kb = hiHalf * 8;
#pragma unroll
        for (int nt = 0; nt < 2; ++nt) {
            const float* wrow = w_off + (size_t)(n + nt * 16) * 64;
#pragma unroll
            for (int ks = 0; ks < 2; ++ks) {
                AFrag t;
#pragma unroll
                for (int e = 0; e < 16; ++e)
                    t.h[e] = (_Float16)wrow[ks * 32 + kb + ((e >= 8) ? 16 : 0) + (e & 7)];
                bf[ks][nt] = t.v;
            }
        }
    }

    // ---- A fragments: A[m][k] = x[b, k, pixel(Mbase+m)] ----
    // Layout (16-bit A, 16x32): M = lane&15,
    //   element e = 2v+j -> K = 16*(v>=4) + 8*(lane>=16) + 2*(v&3) + j
    const int   m   = lane & 15;
    const float* xb = x + (size_t)b * C_ * HW + (pbase + m);
    v16h af[2];
#pragma unroll
    for (int ks = 0; ks < 2; ++ks) {
        const int kbase = ks * 32 + hiHalf * 8;
        AFrag t;
#pragma unroll
        for (int e = 0; e < 16; ++e) {
            const int v = e >> 1, j = e & 1;
            const int K = kbase + ((v >= 4) ? 16 : 0) + 2 * (v & 3) + j;
            t.h[e] = (_Float16)xb[(size_t)K * HW];
        }
        af[ks] = t.v;
    }

    // ---- WMMA accumulate: D = A x B + C ----
    v8f c0 = {};   // out channels 0..15
    v8f c1 = {};   // out channels 16..31
#pragma unroll
    for (int ks = 0; ks < 2; ++ks) {
        c0 = __builtin_amdgcn_wmma_f32_16x16x32_f16(
                 false, af[ks], false, bf[ks][0], (short)0, c0, false, false);
        c1 = __builtin_amdgcn_wmma_f32_16x16x32_f16(
                 false, af[ks], false, bf[ks][1], (short)0, c1, false, false);
    }

    // ---- Store D (+bias). Layout: N = lane&15, M = r + 8*(lane>=16) ----
    const int n0 = lane & 15;
    const int n1 = n0 + 16;
    const float bias0 = b_off[n0];
    const float bias1 = b_off[n1];
    CFrag d0, d1;
    d0.v = c0; d1.v = c1;
    float* offb = off + (size_t)b * OC * HW;
#pragma unroll
    for (int r = 0; r < 8; ++r) {
        const int p = pbase + hiHalf * 8 + r;
        offb[(size_t)n0 * HW + p] = d0.f[r] + bias0;
        offb[(size_t)n1 * HW + p] = d1.f[r] + bias1;
    }
}

// ---------------------------------------------------------------------------
// Kernel 2: bilinear grid-sample with border clamp.
// Simplification of the reference coordinate algebra:
//   ix = clip(w + 0.25*off_x + (sj ? +0.25 : -0.25), 0, W-1)
//   iy = clip(h + 0.25*off_y + (si ? +0.25 : -0.25), 0, H-1)
// where off_x = off[b, gi*4+q, h, w], off_y = off[b, 16+gi*4+q, h, w],
// q = si*2+sj, si = oh&1, sj = ow&1.
// One thread per (b, gi, oh, ow); loops over the group's 16 channels so the
// 4 bilinear taps' address math is amortized. Lanes walk ow -> coalesced.
// ---------------------------------------------------------------------------
__global__ __launch_bounds__(256)
void dysample_sample(const float* __restrict__ x,
                     const float* __restrict__ off,
                     float* __restrict__ out)
{
    const int tid = blockIdx.x * 256 + threadIdx.x;
    const int ow  = tid & (OW_ - 1);
    const int oh  = (tid >> 8) & (OH_ - 1);
    const int bg  = tid >> 16;          // 0..63
    const int gi  = bg & 3;
    const int b   = bg >> 2;

    const int sj = ow & 1, si = oh & 1;
    const int w  = ow >> 1, h  = oh >> 1;
    const int q  = si * 2 + sj;

    const size_t obase = (size_t)b * OC * HW + (size_t)h * W_ + w;
    const float off_x = off[obase + (size_t)(gi * 4 + q) * HW];
    const float off_y = off[obase + (size_t)(16 + gi * 4 + q) * HW];

    const float ix = fminf(fmaxf((float)w + 0.25f * off_x + (sj ? 0.25f : -0.25f),
                                 0.0f), (float)(W_ - 1));
    const float iy = fminf(fmaxf((float)h + 0.25f * off_y + (si ? 0.25f : -0.25f),
                                 0.0f), (float)(H_ - 1));

    const float x0f = floorf(ix), y0f = floorf(iy);
    const float wx = ix - x0f, wy = iy - y0f;
    int x0 = (int)x0f; x0 = (x0 < 0) ? 0 : ((x0 > W_ - 1) ? W_ - 1 : x0);
    int y0 = (int)y0f; y0 = (y0 < 0) ? 0 : ((y0 > H_ - 1) ? H_ - 1 : y0);
    const int x1 = (x0 + 1 > W_ - 1) ? W_ - 1 : x0 + 1;
    const int y1 = (y0 + 1 > H_ - 1) ? H_ - 1 : y0 + 1;

    const float w00 = (1.0f - wy) * (1.0f - wx);
    const float w01 = (1.0f - wy) * wx;
    const float w10 = wy * (1.0f - wx);
    const float w11 = wy * wx;

    const int i00 = y0 * W_ + x0, i01 = y0 * W_ + x1;
    const int i10 = y1 * W_ + x0, i11 = y1 * W_ + x1;

    const float* xg = x   + ((size_t)b * C_ + gi * 16) * HW;
    float*       og = out + ((size_t)b * C_ + gi * 16) * OHW
                          + (size_t)oh * OW_ + ow;
#pragma unroll
    for (int c = 0; c < 16; ++c) {
        const float* xp = xg + (size_t)c * HW;
        og[(size_t)c * OHW] =
            w00 * xp[i00] + w01 * xp[i01] + w10 * xp[i10] + w11 * xp[i11];
    }
}

// ---------------------------------------------------------------------------
extern "C" void kernel_launch(void* const* d_in, const int* in_sizes, int n_in,
                              void* d_out, int out_size, void* d_ws, size_t ws_size,
                              hipStream_t stream)
{
    (void)in_sizes; (void)n_in; (void)out_size; (void)ws_size;
    const float* x     = (const float*)d_in[0];   // (16,64,128,128)
    const float* w_off = (const float*)d_in[1];   // (32,64)
    const float* b_off = (const float*)d_in[2];   // (32,)
    float* off = (float*)d_ws;                    // (16,32,128,128) = 32 MB scratch
    float* out = (float*)d_out;                   // (16,64,256,256)

    // Kernel 1: 262144 pixels / 16 per wave-tile = 16384 waves; 8 waves/block.
    dysample_offset_wmma<<<2048, 256, 0, stream>>>(x, w_off, b_off, off);

    // Kernel 2: 16*4*256*256 = 4,194,304 threads.
    dysample_sample<<<16384, 256, 0, stream>>>(x, off, out);
}
